// MultiHeadAttention_13116830122143
// MI455X (gfx1250) — matrix operand order repaired and checked by reading, not compiled, not run on hardware
//
#include <hip/hip_runtime.h>
#include <hip/hip_bf16.h>
#include <stdint.h>

// MHA forward for B=2,S=2048,D=1024,H=16,DK=64 on gfx1250 (CDNA5, wave32).
// d_out = [out fp32 (B*S*D)] ++ [attn fp32 (B*H*S*S)].
// Workspace: Q,K ([B,H,S,DK]), V transposed ([B,H,DK,S]), O ([B,S,D]) in f16.

typedef __attribute__((ext_vector_type(16))) _Float16 v16h;
typedef __attribute__((ext_vector_type(8)))  _Float16 v8h;
typedef __attribute__((ext_vector_type(8)))  float    v8f;

#define B_  2
#define S_  2048
#define D_  1024
#define H_  16
#define DK_ 64

// ---- WMMA fragment helpers (layouts per CDNA5 ISA 7.12.2, 16-bit data) ----
// A 16x32 (MxK): lane L holds row M=L&15; K elements [hi*8,+8) and
// [16+hi*8,+8), hi=L>>4.  p points at (row, k0 + hi*8).
__device__ __forceinline__ v16h frag_a16(const _Float16* p) {
  v8h lo = *(const v8h*)(p);
  v8h up = *(const v8h*)(p + 16);
  return __builtin_shufflevector(lo, up, 0,1,2,3,4,5,6,7,8,9,10,11,12,13,14,15);
}
// B 32x16 (KxN): lane L holds col N=L&15; K elements [hi*8,+8) and
// [16+hi*8,+8), hi=L>>4.  p points at (k0 + hi*8, col).
__device__ __forceinline__ v16h frag_b16(const _Float16* p) {
  v8h lo = *(const v8h*)(p);
  v8h up = *(const v8h*)(p + 16);
  return __builtin_shufflevector(lo, up, 0,1,2,3,4,5,6,7,8,9,10,11,12,13,14,15);
}
__device__ __forceinline__ v8f wmma_f16(v16h a, v16h b, v8f c) {
  return __builtin_amdgcn_wmma_f32_16x16x32_f16(false, a, false, b, (short)0, c,
                                                false, false);
}
__device__ __forceinline__ v8h cvt8(float4 a, float4 b) {
  v8h p = { (_Float16)a.x, (_Float16)a.y, (_Float16)a.z, (_Float16)a.w,
            (_Float16)b.x, (_Float16)b.y, (_Float16)b.z, (_Float16)b.w };
  return p;
}

// ------------------------------------------------------------------
// Kernel 1: Y[n,o] = scale * sum_d X[n,d] * W[o,d]   (f32 in, f16 out)
// VT=0: Y layout [B,H,S,DK] (Q,K).  VT=1: Y layout [B,H,DK,S] (V transposed,
// per-lane v8h contiguous stores).  Block: 256 thr (8 waves); tile 64x128;
// per K-step each wave does 4 WMMAs (4 M-subtiles sharing one B fragment).
// ------------------------------------------------------------------
template <int VT>
__global__ __launch_bounds__(256) void proj_qkv_kernel(
    const float* __restrict__ X, const float* __restrict__ W,
    _Float16* __restrict__ Y, float scale)
{
  __shared__ _Float16 As[64][40];   // 40-half stride: 16B aligned, bank-spread
  __shared__ _Float16 Bs[128][40];

  const int t    = threadIdx.x;
  const int lane = t & 31, wave = t >> 5;
  const int m    = lane & 15, hf = lane >> 4;
  const int m0   = blockIdx.x * 64;
  const int n0   = blockIdx.y * 128;

  v8f acc0 = {}, acc1 = {}, acc2 = {}, acc3 = {};
  for (int k0 = 0; k0 < D_; k0 += 32) {
    {  // stage A 64x32: 8 floats/thread -> one v8h
      int row = t >> 2, seg = (t & 3) * 8;
      const float4* src = (const float4*)(X + (size_t)(m0 + row) * D_ + k0 + seg);
      *(v8h*)&As[row][seg] = cvt8(src[0], src[1]);
      if (k0 + 32 < D_)
        __builtin_prefetch(X + (size_t)(m0 + row) * D_ + k0 + 32 + seg, 0, 3);
    }
    {  // stage B 128x32: Bs[o_local][k] = W[n0+o][k0+k], 16 floats/thread
      int row = t >> 1, h16 = (t & 1) * 16;
      const float4* src = (const float4*)(W + (size_t)(n0 + row) * D_ + k0 + h16);
      *(v8h*)&Bs[row][h16]     = cvt8(src[0], src[1]);
      *(v8h*)&Bs[row][h16 + 8] = cvt8(src[2], src[3]);
      if (k0 + 32 < D_)
        __builtin_prefetch(W + (size_t)(n0 + row) * D_ + k0 + 32 + h16, 0, 3);
    }
    __syncthreads();
    v16h b  = frag_b16(&Bs[wave * 16 + m][hf * 8]);
    v16h a0 = frag_a16(&As[ 0 + m][hf * 8]);
    v16h a1 = frag_a16(&As[16 + m][hf * 8]);
    v16h a2 = frag_a16(&As[32 + m][hf * 8]);
    v16h a3 = frag_a16(&As[48 + m][hf * 8]);
    acc0 = wmma_f16(a0, b, acc0);
    acc1 = wmma_f16(a1, b, acc1);
    acc2 = wmma_f16(a2, b, acc2);
    acc3 = wmma_f16(a3, b, acc3);
    __syncthreads();
  }

  const int o = n0 + wave * 16 + m;       // output column
  const int h = o >> 6, dk = o & 63;
  auto store_tile = [&](const v8f& acc, int ms) {
    if (VT) {
      // [B,H,DK,S]: 8 consecutive s per lane -> one b128 store
      int n = m0 + ms + hf * 8;           // first row (= b*S + s)
      int b = n >> 11, s = n & (S_ - 1);
      v8h p;
      #pragma unroll
      for (int r = 0; r < 8; ++r) p[r] = (_Float16)(acc[r] * scale);
      *(v8h*)&Y[(((size_t)(b * H_ + h)) * DK_ + dk) * S_ + s] = p;
    } else {
      #pragma unroll
      for (int r = 0; r < 8; ++r) {
        int n = m0 + ms + r + hf * 8;
        int b = n >> 11, s = n & (S_ - 1);
        Y[(((size_t)(b * H_ + h)) * S_ + s) * DK_ + dk] = (_Float16)(acc[r] * scale);
      }
    }
  };
  store_tile(acc0, 0); store_tile(acc1, 16);
  store_tile(acc2, 32); store_tile(acc3, 48);
}

// ------------------------------------------------------------------
// Kernel 2: per (b,h,16-row q tile): WMMA scores -> causal softmax
// (probs compacted to f16 in-place in LDS) -> NT-stream attn rows ->
// WMMA P@V with transposed V (8 waves, K-split + LDS reduce).
// Dynamic LDS: 16x2048 f32 scores (reused as f16 probs) + Q tile + reduce.
// ------------------------------------------------------------------
__global__ __launch_bounds__(256) void attn_kernel(
    const _Float16* __restrict__ Qw, const _Float16* __restrict__ Kw,
    const _Float16* __restrict__ Vt, float* __restrict__ attn,
    _Float16* __restrict__ Ow)
{
  extern __shared__ char smem[];
  float*    Ps = (float*)smem;                                   // [16][S_] f32
  _Float16* Ph = (_Float16*)smem;                                // [16][S_] f16
  // Ph row R uses bytes [R*8192, R*8192+4096) = first half of f32 row R.
  _Float16* Qs = (_Float16*)(smem + (size_t)16 * S_ * 4);        // [16][DK_]
  float*    Rs = (float*)(smem + (size_t)16 * S_ * 4
                               + (size_t)16 * DK_ * 2);          // [4][16][16]
  const int PHS = 2 * S_;   // Ph row stride in halves (= one f32 row)

  const int t    = threadIdx.x;
  const int lane = t & 31, wave = t >> 5;
  const int m    = lane & 15, hf = lane >> 4;
  const int qt = blockIdx.x, h = blockIdx.y, b = blockIdx.z;
  const int q0 = qt * 16;
  const int bh = b * H_ + h;

  const _Float16* Qb  = Qw + (size_t)bh * S_ * DK_;
  const _Float16* Kb  = Kw + (size_t)bh * S_ * DK_;
  const _Float16* Vtb = Vt + (size_t)bh * DK_ * S_;
  float* attnb = attn + ((size_t)bh * S_ + q0) * S_;

  {  // load Q tile: 1024 halves, 4 per thread (8B)
    int idx = t * 4;
    *(uint2*)&Qs[idx] = *(const uint2*)(Qb + (size_t)q0 * DK_ + idx);
  }
  __syncthreads();

  // Q fragments for DK=64 (two 32-wide K steps), reused across all key tiles
  v16h aq0 = frag_a16(&Qs[m * DK_ + 0  + hf * 8]);
  v16h aq1 = frag_a16(&Qs[m * DK_ + 32 + hf * 8]);

  // ---- scores: causal prefix only (kt <= qt), round-robin over 8 waves ----
  const int nk = qt + 1;
  for (int kt = wave; kt < nk; kt += 8) {
    const int kc0 = kt * 16;
    if (kt + 8 < nk)
      __builtin_prefetch(Kb + (size_t)(kc0 + 128 + m) * DK_ + hf * 16, 0, 3);
    v8f c = {};
    v16h b0 = frag_b16(Kb + (size_t)(kc0 + m) * DK_ + 0  + hf * 8);
    c = wmma_f16(aq0, b0, c);
    v16h b1 = frag_b16(Kb + (size_t)(kc0 + m) * DK_ + 32 + hf * 8);
    c = wmma_f16(aq1, b1, c);
    const int kcol = kc0 + m;
    #pragma unroll
    for (int r = 0; r < 8; ++r) {
      int qrow = r + hf * 8;
      float v = c[r];
      if (kcol > q0 + qrow) v = -__builtin_inff();   // causal mask
      Ps[qrow * S_ + kcol] = v;
    }
  }
  __syncthreads();

  const int kend = nk * 16;
  const int kpad = (kend + 31) & ~31;
  // ---- softmax: 2 rows per wave; probs -> f16 LDS + NT f32 global ----
  #pragma unroll
  for (int rr = 0; rr < 2; ++rr) {
    const int row = wave * 2 + rr;
    float*    pr = Ps + row * S_;
    _Float16* ph = Ph + row * PHS;
    float mx = -__builtin_inff();
    for (int c = lane; c < kend; c += 32) mx = fmaxf(mx, pr[c]);
    #pragma unroll
    for (int off = 16; off; off >>= 1) mx = fmaxf(mx, __shfl_xor(mx, off));
    float sum = 0.f;
    for (int c = lane; c < kend; c += 32) {
      float e = __expf(pr[c] - mx); pr[c] = e; sum += e;
    }
    #pragma unroll
    for (int off = 16; off; off >>= 1) sum += __shfl_xor(sum, off);
    const float inv = 1.0f / sum;
    float* arow = attnb + (size_t)row * S_;
    // In-place f16 compaction: write at byte 2c overlaps f32 element c/2 of
    // the SAME row, which every lane has already consumed (reads of iteration
    // i complete before its stores issue; clobbered f32 index c/2 < c).
    for (int c = lane; c < kend; c += 32) {
      float p = pr[c] * inv;
      ph[c] = (_Float16)p;
      __builtin_nontemporal_store(p, &arow[c]);          // stream 537MB attn
    }
    for (int c = kend + lane; c < S_; c += 32)
      __builtin_nontemporal_store(0.f, &arow[c]);        // masked tail
    for (int c = kend + lane; c < kpad; c += 32)
      ph[c] = (_Float16)0.f;                             // pad for WMMA
  }
  __syncthreads();

  // ---- out tile [16,64] = P[16,kend] @ V[kend,64] ----
  // 8 waves: (wave&3) -> DK n-tile, (wave>>2) -> K half; reduce via LDS.
  // A = f16 probs from LDS (b128 loads); B = transposed V (contiguous K).
  const int nt = (wave & 3) * 16;
  const int kh = wave >> 2;
  const _Float16* vrow = Vtb + (size_t)(nt + m) * S_;
  v8f c = {};
  for (int k0 = kh * 32; k0 < kpad; k0 += 64) {
    v16h ap = frag_a16(Ph + m * PHS + k0 + hf * 8);
    v16h bv = frag_b16(vrow + k0 + hf * 8);
    c = wmma_f16(ap, bv, c);
  }
  if (kh == 1) {
    #pragma unroll
    for (int r = 0; r < 8; ++r)
      Rs[(wave & 3) * 256 + (r + hf * 8) * 16 + m] = c[r];
  }
  __syncthreads();
  if (kh == 0) {
    #pragma unroll
    for (int r = 0; r < 8; ++r) {
      float v = c[r] + Rs[(wave & 3) * 256 + (r + hf * 8) * 16 + m];
      int qrow = q0 + r + hf * 8;
      Ow[((size_t)b * S_ + qrow) * D_ + h * DK_ + nt + m] = (_Float16)v;
    }
  }
}

// ------------------------------------------------------------------
// Kernel 3: out[n,o] = sum_d O[n,d]*Wo[o,d] + bo[o]   (f16 A, f32 W/out)
// Same 64x128 tiling as kernel 1.
// ------------------------------------------------------------------
__global__ __launch_bounds__(256) void out_proj_kernel(
    const _Float16* __restrict__ A, const float* __restrict__ W,
    const float* __restrict__ bias, float* __restrict__ Y)
{
  __shared__ _Float16 As[64][40];
  __shared__ _Float16 Bs[128][40];

  const int t    = threadIdx.x;
  const int lane = t & 31, wave = t >> 5;
  const int m    = lane & 15, hf = lane >> 4;
  const int m0   = blockIdx.x * 64;
  const int n0   = blockIdx.y * 128;

  v8f acc0 = {}, acc1 = {}, acc2 = {}, acc3 = {};
  for (int k0 = 0; k0 < D_; k0 += 32) {
    {  // stage A (already f16): 8 halves/thread, one b128 copy
      int row = t >> 2, seg = (t & 3) * 8;
      *(v8h*)&As[row][seg] =
          *(const v8h*)(A + (size_t)(m0 + row) * D_ + k0 + seg);
      if (k0 + 32 < D_)
        __builtin_prefetch(A + (size_t)(m0 + row) * D_ + k0 + 32 + seg, 0, 3);
    }
    {  // stage B from Wo f32
      int row = t >> 1, h16 = (t & 1) * 16;
      const float4* src = (const float4*)(W + (size_t)(n0 + row) * D_ + k0 + h16);
      *(v8h*)&Bs[row][h16]     = cvt8(src[0], src[1]);
      *(v8h*)&Bs[row][h16 + 8] = cvt8(src[2], src[3]);
      if (k0 + 32 < D_)
        __builtin_prefetch(W + (size_t)(n0 + row) * D_ + k0 + 32 + h16, 0, 3);
    }
    __syncthreads();
    v16h b  = frag_b16(&Bs[wave * 16 + m][hf * 8]);
    v16h a0 = frag_a16(&As[ 0 + m][hf * 8]);
    v16h a1 = frag_a16(&As[16 + m][hf * 8]);
    v16h a2 = frag_a16(&As[32 + m][hf * 8]);
    v16h a3 = frag_a16(&As[48 + m][hf * 8]);
    acc0 = wmma_f16(a0, b, acc0);
    acc1 = wmma_f16(a1, b, acc1);
    acc2 = wmma_f16(a2, b, acc2);
    acc3 = wmma_f16(a3, b, acc3);
    __syncthreads();
  }

  const int col = n0 + wave * 16 + m;
  const float bb = bias[col];
  auto store_tile = [&](const v8f& acc, int ms) {
    #pragma unroll
    for (int r = 0; r < 8; ++r) {
      int row = m0 + ms + r + hf * 8;
      Y[(size_t)row * D_ + col] = acc[r] + bb;
    }
  };
  store_tile(acc0, 0); store_tile(acc1, 16);
  store_tile(acc2, 32); store_tile(acc3, 48);
}

// ------------------------------------------------------------------
extern "C" void kernel_launch(void* const* d_in, const int* in_sizes, int n_in,
                              void* d_out, int out_size, void* d_ws, size_t ws_size,
                              hipStream_t stream) {
  (void)in_sizes; (void)n_in; (void)out_size; (void)ws_size;

  const float* query = (const float*)d_in[0];
  const float* key_  = (const float*)d_in[1];
  const float* value = (const float*)d_in[2];
  // d_in[3] = causal mask (int32) — causality is hardcoded in attn_kernel
  const float* wq = (const float*)d_in[4];
  const float* wk = (const float*)d_in[5];
  const float* wv = (const float*)d_in[6];
  const float* wo = (const float*)d_in[7];
  const float* bo = (const float*)d_in[8];

  float* out  = (float*)d_out;
  float* attn = out + (size_t)B_ * S_ * D_;

  const size_t qkv_elems = (size_t)B_ * H_ * S_ * DK_;   // 4M halves each
  _Float16* Qw = (_Float16*)d_ws;
  _Float16* Kw = Qw + qkv_elems;
  _Float16* Vt = Kw + qkv_elems;                         // [B,H,DK,S] f16
  _Float16* Ow = Vt + qkv_elems;                         // [B,S,D] f16

  dim3 blk(256);
  dim3 gproj((B_ * S_) / 64, D_ / 128);                  // (64, 8)

  proj_qkv_kernel<0><<<gproj, blk, 0, stream>>>(query, wq, Qw, 0.125f); // 1/sqrt(DK)
  proj_qkv_kernel<0><<<gproj, blk, 0, stream>>>(key_,  wk, Kw, 1.0f);
  proj_qkv_kernel<1><<<gproj, blk, 0, stream>>>(value, wv, Vt, 1.0f);

  const size_t smem = (size_t)16 * S_ * sizeof(float)        // scores/probs
                    + (size_t)16 * DK_ * sizeof(_Float16)    // Q tile
                    + (size_t)4 * 16 * 16 * sizeof(float);   // PV reduce
  (void)hipFuncSetAttribute((const void*)attn_kernel,
                            hipFuncAttributeMaxDynamicSharedMemorySize,
                            (int)smem);
  attn_kernel<<<dim3(S_ / 16, H_, B_), blk, smem, stream>>>(Qw, Kw, Vt, attn, Ow);

  out_proj_kernel<<<gproj, blk, 0, stream>>>(Ow, wo, bo, out);
}
